// MotionVAE_18348100288743
// MI455X (gfx1250) — matrix operand order repaired and checked by reading, not compiled, not run on hardware
//
#include <hip/hip_runtime.h>
#include <hip/hip_bf16.h>
#include <stdint.h>

// ---------------------------------------------------------------------------
// MotionVAE forward for MI455X (gfx1250), wave32 + WMMA f16 (16x16x32, f32 acc)
// Compute-bound (~65 GFLOP vs ~60MB HBM traffic) -> all GEMMs run on
// v_wmma_f32_16x16x32_f16. Fragment loads exploit CDNA5 layout contiguity:
// A = 2x global_load_b128 per lane; B staged transposed in LDS (stride 40
// halfs, conflict-free) -> 2x ds_load_b128 per lane. Staging is
// register-buffered + double-buffered: the next tile's global load is issued
// before the current WMMA and committed to LDS after it, so its latency hides
// under the matrix op. k_gemm uses two accumulators to break the WMMA RAW
// chain. VQ argmins (both directions) fused via packed 64-bit min-keys.
// ---------------------------------------------------------------------------

typedef __attribute__((ext_vector_type(16))) _Float16 v16h;
typedef __attribute__((ext_vector_type(8)))  _Float16 h8;
typedef __attribute__((ext_vector_type(4)))  _Float16 h4;
typedef __attribute__((ext_vector_type(8)))  float    v8f;

#define NB   32768   // batch
#define MCB  2048    // codebook size
#define BT_STRIDE 40 // padded LDS row stride (halfs): b128 reads conflict-free
#define BT_TILE   (16 * BT_STRIDE)

// ------------------------------ WMMA helpers -------------------------------

__device__ __forceinline__ v8f wmma_f16(v16h a, v16h b, v8f c) {
  // (neg_a, A, neg_b, B, c_mod, C, reuse_a, reuse_b)
  return __builtin_amdgcn_wmma_f32_16x16x32_f16(false, a, false, b, (short)0, c,
                                                false, false);
}

__device__ __forceinline__ v16h cat8(h8 lo, h8 hi) {
  return __builtin_shufflevector(lo, hi, 0, 1, 2, 3, 4, 5, 6, 7,
                                 8, 9, 10, 11, 12, 13, 14, 15);
}

// A fragment: 16x32 f16 tile, row-major, stride lda (halfs, multiple of 16).
// Per-lane elements 0..7 <- K=kh..kh+7, 8..15 <- K=16+kh..16+kh+7 (kh=0|8):
// two contiguous 16B loads.
__device__ __forceinline__ v16h load_a_frag(const _Float16* __restrict__ A,
                                            int lda) {
  const int lane = threadIdx.x & 31;
  const int m = lane & 15;
  const int kh = (lane & 16) ? 8 : 0;
  const _Float16* r = A + (size_t)m * lda + kh;
  h8 lo = *(const h8*)(r);
  h8 hi = *(const h8*)(r + 16);
  return cat8(lo, hi);
}

// B fragment from transposed LDS tile Bt[n][k], row stride BT_STRIDE halfs.
// Per-lane elements 0..7 <- K=khB..khB+7, 8..15 <- K=16+khB.. (khB=0|8) at fixed column n.
__device__ __forceinline__ v16h load_b_frag_t(const _Float16* __restrict__ shBt) {
  const int lane = threadIdx.x & 31;
  const int n = lane & 15;
  const int khB = (lane & 16) ? 8 : 0;
  const _Float16* p = shBt + n * BT_STRIDE + khB;
  h8 lo = *(const h8*)(p);
  h8 hi = *(const h8*)(p + 16);
  return cat8(lo, hi);
}

// Register-buffered transpose staging of a 32(K)x16(N) global tile into
// Bt[n][k]: load (coalesced b64) and deferred LDS commit are split so the
// global load can fly under the current WMMA.
__device__ __forceinline__ h4 stage_b_load(const _Float16* __restrict__ Bw,
                                           int ldb, int kk, int col0) {
  const int t = threadIdx.x;
  const int k = t >> 2;
  const int n0 = (t & 3) << 2;
  return *(const h4*)(Bw + (size_t)(kk + k) * ldb + col0 + n0);
}

__device__ __forceinline__ void stage_b_store(h4 v,
                                              _Float16* __restrict__ shBt) {
  const int t = threadIdx.x;
  const int k = t >> 2;
  const int n0 = (t & 3) << 2;
#pragma unroll
  for (int i = 0; i < 4; ++i) shBt[(n0 + i) * BT_STRIDE + k] = v[i];
}

__device__ __forceinline__ float elu1(float x) {
  return x > 0.0f ? x : expm1f(x);
}

// ------------------------- generic WMMA GEMM kernel ------------------------
// C[M,N] = act(A[M,K] @ B[K,Npad] + bias), block = 4 waves, tile 64x16.
__global__ void k_gemm(const _Float16* __restrict__ A, int lda,
                       const _Float16* __restrict__ Bw, int ldb,
                       const float* __restrict__ bias,
                       _Float16* __restrict__ outH, int ldh,
                       float* __restrict__ outF, int ldf,
                       int K, int n_valid, int act) {
  __shared__ __align__(16) _Float16 shBt[2][BT_TILE];
  const int tid = threadIdx.x;
  const int lane = tid & 31, wave = tid >> 5;
  const int row0 = blockIdx.y * 64 + wave * 16;
  const int col0 = blockIdx.x * 16;
  const _Float16* Arow = A + (size_t)row0 * lda;
  const int nsteps = K >> 5;
  v8f acc0 = {}, acc1 = {};

  stage_b_store(stage_b_load(Bw, ldb, 0, col0), shBt[0]);
  __syncthreads();
  for (int s = 0; s < nsteps; s += 2) {
    const bool have1 = (s + 1 < nsteps);
    const bool have2 = (s + 2 < nsteps);
    // even step: compute from shBt[0]; preload s+1, commit after the WMMA.
    {
      v16h a = load_a_frag(Arow + (s << 5), lda);
      h4 pf = {};
      if (have1) pf = stage_b_load(Bw, ldb, (s + 1) << 5, col0);
      if (have2) __builtin_prefetch(Arow + ((s + 2) << 5), 0, 1);
      acc0 = wmma_f16(a, load_b_frag_t(shBt[0]), acc0);
      if (have1) stage_b_store(pf, shBt[1]);
    }
    __syncthreads();
    if (have1) {
      // odd step: compute from shBt[1]; preload s+2, commit after the WMMA.
      v16h a = load_a_frag(Arow + ((s + 1) << 5), lda);
      h4 pf = {};
      if (have2) pf = stage_b_load(Bw, ldb, (s + 2) << 5, col0);
      acc1 = wmma_f16(a, load_b_frag_t(shBt[1]), acc1);
      if (have2) stage_b_store(pf, shBt[0]);
      __syncthreads();
    }
  }
  v8f acc = acc0 + acc1;

  const int col = col0 + (lane & 15);
  const int rbase = row0 + ((lane & 16) ? 8 : 0);
  if (col < n_valid) {
    const float bv = bias ? bias[col] : 0.0f;
#pragma unroll
    for (int v = 0; v < 8; ++v) {
      const int row = rbase + v;
      float x = acc[v] + bv;
      if (act) x = elu1(x);
      if (outH) outH[(size_t)row * ldh + col] = (_Float16)x;
      if (outF) outF[(size_t)row * ldf + col] = x;
    }
  }
}

// --------------------- blended 4-expert WMMA GEMM kernel -------------------
// out = act( sum_e g[n,e] * (A @ W_e + b_e) ); 4 accumulators (distance-4
// WMMA chains) + register-buffered double-buffered 4-tile staging.
__global__ void k_gemm_expert(const _Float16* __restrict__ A, int lda,
                              const _Float16* __restrict__ Bw, int ldb,
                              int wstride,
                              const float* __restrict__ g,
                              const float* __restrict__ bias, int bias_stride,
                              _Float16* __restrict__ outH, int ldh,
                              float* __restrict__ outF, int ldf,
                              int K, int n_valid, int act) {
  __shared__ __align__(16) _Float16 shBt[2][4 * BT_TILE];
  const int tid = threadIdx.x;
  const int lane = tid & 31, wave = tid >> 5;
  const int row0 = blockIdx.y * 64 + wave * 16;
  const int col0 = blockIdx.x * 16;
  const _Float16* Arow = A + (size_t)row0 * lda;
  const int nsteps = K >> 5;
  v8f acc0 = {}, acc1 = {}, acc2 = {}, acc3 = {};

#pragma unroll
  for (int e = 0; e < 4; ++e)
    stage_b_store(stage_b_load(Bw + (size_t)e * wstride, ldb, 0, col0),
                  shBt[0] + e * BT_TILE);
  __syncthreads();
  for (int s = 0; s < nsteps; s += 2) {
    const bool have1 = (s + 1 < nsteps);
    const bool have2 = (s + 2 < nsteps);
    {
      v16h a = load_a_frag(Arow + (s << 5), lda);
      h4 pf[4] = {{}, {}, {}, {}};
      if (have1) {
#pragma unroll
        for (int e = 0; e < 4; ++e)
          pf[e] = stage_b_load(Bw + (size_t)e * wstride, ldb, (s + 1) << 5, col0);
      }
      if (have2) __builtin_prefetch(Arow + ((s + 2) << 5), 0, 1);
      acc0 = wmma_f16(a, load_b_frag_t(shBt[0] + 0 * BT_TILE), acc0);
      acc1 = wmma_f16(a, load_b_frag_t(shBt[0] + 1 * BT_TILE), acc1);
      acc2 = wmma_f16(a, load_b_frag_t(shBt[0] + 2 * BT_TILE), acc2);
      acc3 = wmma_f16(a, load_b_frag_t(shBt[0] + 3 * BT_TILE), acc3);
      if (have1) {
#pragma unroll
        for (int e = 0; e < 4; ++e) stage_b_store(pf[e], shBt[1] + e * BT_TILE);
      }
    }
    __syncthreads();
    if (have1) {
      v16h a = load_a_frag(Arow + ((s + 1) << 5), lda);
      h4 pf[4] = {{}, {}, {}, {}};
      if (have2) {
#pragma unroll
        for (int e = 0; e < 4; ++e)
          pf[e] = stage_b_load(Bw + (size_t)e * wstride, ldb, (s + 2) << 5, col0);
      }
      acc0 = wmma_f16(a, load_b_frag_t(shBt[1] + 0 * BT_TILE), acc0);
      acc1 = wmma_f16(a, load_b_frag_t(shBt[1] + 1 * BT_TILE), acc1);
      acc2 = wmma_f16(a, load_b_frag_t(shBt[1] + 2 * BT_TILE), acc2);
      acc3 = wmma_f16(a, load_b_frag_t(shBt[1] + 3 * BT_TILE), acc3);
      if (have2) {
#pragma unroll
        for (int e = 0; e < 4; ++e) stage_b_store(pf[e], shBt[0] + e * BT_TILE);
      }
      __syncthreads();
    }
  }

  const int col = col0 + (lane & 15);
  const int rbase = row0 + ((lane & 16) ? 8 : 0);
  if (col < n_valid) {
    const float b0 = bias[0 * bias_stride + col];
    const float b1 = bias[1 * bias_stride + col];
    const float b2 = bias[2 * bias_stride + col];
    const float b3 = bias[3 * bias_stride + col];
#pragma unroll
    for (int v = 0; v < 8; ++v) {
      const int row = rbase + v;
      const float* gr = g + (size_t)row * 4;
      float x = gr[0] * (acc0[v] + b0) + gr[1] * (acc1[v] + b1) +
                gr[2] * (acc2[v] + b2) + gr[3] * (acc3[v] + b3);
      if (act) x = elu1(x);
      if (outH) outH[(size_t)row * ldh + col] = (_Float16)x;
      if (outF) outF[(size_t)row * ldf + col] = x;
    }
  }
}

// ------------------------------ fused VQ kernel ----------------------------
// d2[n,m] = |x_n|^2 + |y_m|^2 - 2 x.y via WMMA; both argmins via 64-bit
// min-keys (d2 bits << 32 | index): lowest index wins ties, matching argmin.
// shCB stored [code][k] padded to BT_STRIDE: fragment reads are 2x b128.
__global__ void k_vq(const _Float16* __restrict__ LATh,
                     const _Float16* __restrict__ CBh,
                     const float* __restrict__ xnorm,
                     const float* __restrict__ ynorm,
                     unsigned long long* __restrict__ rowkey,
                     unsigned long long* __restrict__ colkey) {
  __shared__ __align__(16) _Float16 shCB[64 * BT_STRIDE];
  __shared__ float shY[64];
  __shared__ unsigned long long shRK[64], shCK[64];
  const int tid = threadIdx.x;
  const int lane = tid & 31, wave = tid >> 5;
  const int c0 = blockIdx.x * 64;
  const int r0blk = blockIdx.y * 64;
  if (tid < 64) {
    shRK[tid] = ~0ull;
    shCK[tid] = ~0ull;
    shY[tid] = ynorm[c0 + tid];
  }
  {  // stage 64x32 codebook tile: thread t -> code t>>1, halfs (t&1)*16..+15
    const int code = tid >> 1;
    const int h0 = (tid & 1) << 4;
    const _Float16* src = CBh + (size_t)(c0 + code) * 32 + h0;
    _Float16* dst = shCB + code * BT_STRIDE + h0;
    *(h8*)(dst) = *(const h8*)(src);
    *(h8*)(dst + 8) = *(const h8*)(src + 8);
  }
  __syncthreads();

  const int row0 = r0blk + wave * 16;
  v16h a = load_a_frag(LATh + (size_t)row0 * 32, 32);
  const int rb = wave * 16 + ((lane & 16) ? 8 : 0);
  float xn[8];
#pragma unroll
  for (int v = 0; v < 8; ++v) xn[v] = xnorm[r0blk + rb + v];

  const int khB = (lane & 16) ? 8 : 0;
#pragma unroll
  for (int cs = 0; cs < 4; ++cs) {
    const int nloc = cs * 16 + (lane & 15);
    const _Float16* p = shCB + nloc * BT_STRIDE + khB;
    v16h b = cat8(*(const h8*)(p), *(const h8*)(p + 16));
    v8f acc = {};
    acc = wmma_f16(a, b, acc);
    const float yn = shY[nloc];
#pragma unroll
    for (int v = 0; v < 8; ++v) {
      const float d2 = fmaxf(xn[v] + yn - 2.0f * acc[v], 0.0f);
      const unsigned long long db = (unsigned long long)__float_as_uint(d2) << 32;
      atomicMin(&shRK[rb + v], db | (unsigned)(c0 + nloc));
      atomicMin(&shCK[nloc], db | (unsigned)(r0blk + rb + v));
    }
  }
  __syncthreads();
  if (tid < 64)
    atomicMin(&rowkey[r0blk + tid], shRK[tid]);
  else if (tid < 128)
    atomicMin(&colkey[c0 + tid - 64], shCK[tid - 64]);
}

// ----------------------------- small utility kernels -----------------------

__global__ void k_pad_convert(const float* __restrict__ src, int Rs, int Cs,
                              _Float16* __restrict__ dst, int Rd, int Cd) {
  const int total = Rd * Cd;
  for (int i = blockIdx.x * blockDim.x + threadIdx.x; i < total;
       i += gridDim.x * blockDim.x) {
    const int r = i / Cd, c = i % Cd;
    const float v = (r < Rs && c < Cs) ? src[(size_t)r * Cs + c] : 0.0f;
    dst[i] = (_Float16)v;
  }
}

__global__ void k_build_x0(const float* __restrict__ obs,
                           const float* __restrict__ nxt,
                           _Float16* __restrict__ X0) {
  const int total = NB * 288;
  for (int i = blockIdx.x * blockDim.x + threadIdx.x; i < total;
       i += gridDim.x * blockDim.x) {
    const int n = i / 288, c = i % 288;
    float v = 0.0f;
    if (c < 132)      v = obs[(size_t)n * 132 + c];
    else if (c < 264) v = nxt[(size_t)n * 132 + (c - 132)];
    X0[i] = (_Float16)v;
  }
}

__global__ void k_row_norm(const float* __restrict__ X, int cols,
                           float* __restrict__ out, int rows) {
  const int r = blockIdx.x * blockDim.x + threadIdx.x;
  if (r >= rows) return;
  const float* p = X + (size_t)r * cols;
  float s = 0.0f;
  for (int c = 0; c < cols; ++c) { const float v = p[c]; s += v * v; }
  out[r] = s;
}

__global__ void k_init_keys(unsigned long long* __restrict__ rowkey, int n,
                            unsigned long long* __restrict__ colkey, int m) {
  for (int i = blockIdx.x * blockDim.x + threadIdx.x; i < n + m;
       i += gridDim.x * blockDim.x) {
    if (i < n) rowkey[i] = ~0ull;
    else       colkey[i - n] = ~0ull;
  }
}

__global__ void k_finalize_rows(const unsigned long long* __restrict__ rowkey,
                                const float* __restrict__ codebook,
                                const float* __restrict__ obs,
                                float* __restrict__ latdec,
                                _Float16* __restrict__ XE) {
  const int n = blockIdx.x * blockDim.x + threadIdx.x;
  if (n >= NB) return;
  const int idx = (int)(rowkey[n] & 0xffffffffull);
  const float* cb = codebook + (size_t)idx * 32;
  _Float16* xe = XE + (size_t)n * 192;
  float* ld = latdec + (size_t)n * 32;
#pragma unroll
  for (int k = 0; k < 32; ++k) { const float v = cb[k]; ld[k] = v; xe[k] = (_Float16)v; }
  const float* ob = obs + (size_t)n * 132;
  for (int c = 0; c < 132; ++c) xe[32 + c] = (_Float16)ob[c];
#pragma unroll
  for (int c = 164; c < 192; ++c) xe[c] = (_Float16)0.0f;
}

__global__ void k_finalize_cols(const unsigned long long* __restrict__ colkey,
                                const float* __restrict__ latent_ori,
                                float* __restrict__ latenc) {
  const int m = blockIdx.x * blockDim.x + threadIdx.x;
  if (m >= MCB) return;
  const int idx = (int)(colkey[m] & 0xffffffffull);
  const float* src = latent_ori + (size_t)idx * 32;
  float* dst = latenc + (size_t)m * 32;
#pragma unroll
  for (int k = 0; k < 32; ++k) dst[k] = src[k];
}

__global__ void k_softmax4(const float* __restrict__ logits,
                           float* __restrict__ g) {
  const int n = blockIdx.x * blockDim.x + threadIdx.x;
  if (n >= NB) return;
  const float l0 = logits[(size_t)n * 4 + 0], l1 = logits[(size_t)n * 4 + 1];
  const float l2 = logits[(size_t)n * 4 + 2], l3 = logits[(size_t)n * 4 + 3];
  const float mx = fmaxf(fmaxf(l0, l1), fmaxf(l2, l3));
  const float e0 = __expf(l0 - mx), e1 = __expf(l1 - mx);
  const float e2 = __expf(l2 - mx), e3 = __expf(l3 - mx);
  const float inv = 1.0f / (e0 + e1 + e2 + e3);
  g[(size_t)n * 4 + 0] = e0 * inv;
  g[(size_t)n * 4 + 1] = e1 * inv;
  g[(size_t)n * 4 + 2] = e2 * inv;
  g[(size_t)n * 4 + 3] = e3 * inv;
}

// --------------------------------- host ------------------------------------

static inline char* carve(char*& p, size_t bytes) {
  char* r = p;
  p += (bytes + 255) & ~(size_t)255;
  return r;
}

extern "C" void kernel_launch(void* const* d_in, const int* in_sizes, int n_in,
                              void* d_out, int out_size, void* d_ws,
                              size_t ws_size, hipStream_t stream) {
  (void)in_sizes; (void)n_in; (void)out_size; (void)ws_size;
  // inputs: dict order of setup_inputs / params
  const float* obs   = (const float*)d_in[0];
  const float* nxt   = (const float*)d_in[1];
  const float* enc_w1 = (const float*)d_in[2];  const float* enc_b1 = (const float*)d_in[3];
  const float* enc_w2 = (const float*)d_in[4];  const float* enc_b2 = (const float*)d_in[5];
  const float* enc_w3 = (const float*)d_in[6];  const float* enc_b3 = (const float*)d_in[7];
  const float* g_w1   = (const float*)d_in[8];  const float* g_b1   = (const float*)d_in[9];
  const float* g_w2   = (const float*)d_in[10]; const float* g_b2   = (const float*)d_in[11];
  const float* g_w3   = (const float*)d_in[12]; const float* g_b3   = (const float*)d_in[13];
  const float* x_w1   = (const float*)d_in[14]; const float* x_b1   = (const float*)d_in[15];
  const float* x_w2   = (const float*)d_in[16]; const float* x_b2   = (const float*)d_in[17];
  const float* x_w3   = (const float*)d_in[18]; const float* x_b3   = (const float*)d_in[19];
  const float* codebook = (const float*)d_in[20];

  // output tuple layout (flat, f32): output[N,132] | lat_ori[N,32] | lat_dec[N,32] | lat_enc[M,32]
  float* out        = (float*)d_out;
  float* out_output = out;
  float* out_lori   = out + (size_t)NB * 132;
  float* out_ldec   = out_lori + (size_t)NB * 32;
  float* out_lenc   = out_ldec + (size_t)NB * 32;

  // workspace carve
  char* p = (char*)d_ws;
  _Float16* X0h  = (_Float16*)carve(p, (size_t)NB * 288 * 2);
  _Float16* bufA = (_Float16*)carve(p, (size_t)NB * 512 * 2);  // H1 / G2 / HE2
  _Float16* bufB = (_Float16*)carve(p, (size_t)NB * 256 * 2);  // H2 / G1 / HE1
  _Float16* LATh = (_Float16*)carve(p, (size_t)NB * 32 * 2);
  _Float16* XEh  = (_Float16*)carve(p, (size_t)NB * 192 * 2);
  float*    logits = (float*)carve(p, (size_t)NB * 4 * 4);
  float*    gbuf   = (float*)carve(p, (size_t)NB * 4 * 4);
  _Float16* W1p  = (_Float16*)carve(p, (size_t)288 * 512 * 2);
  _Float16* W2p  = (_Float16*)carve(p, (size_t)512 * 256 * 2);
  _Float16* W3p  = (_Float16*)carve(p, (size_t)256 * 32 * 2);
  _Float16* GW1p = (_Float16*)carve(p, (size_t)32 * 256 * 2);
  _Float16* GW2p = (_Float16*)carve(p, (size_t)256 * 256 * 2);
  _Float16* GW3p = (_Float16*)carve(p, (size_t)256 * 16 * 2);
  _Float16* XW1p = (_Float16*)carve(p, (size_t)4 * 192 * 256 * 2);
  _Float16* XW2p = (_Float16*)carve(p, (size_t)4 * 256 * 256 * 2);
  _Float16* XW3p = (_Float16*)carve(p, (size_t)4 * 256 * 144 * 2);
  _Float16* CBh  = (_Float16*)carve(p, (size_t)MCB * 32 * 2);
  float* xnorm = (float*)carve(p, (size_t)NB * 4);
  float* ynorm = (float*)carve(p, (size_t)MCB * 4);
  unsigned long long* rowkey = (unsigned long long*)carve(p, (size_t)NB * 8);
  unsigned long long* colkey = (unsigned long long*)carve(p, (size_t)MCB * 8);

  const int TPB = 256;
  auto gsz = [](size_t total, int tpb) {
    size_t b = (total + tpb - 1) / tpb;
    return (unsigned)(b > 4096 ? 4096 : b);
  };

  // ---- weight conversion / padding (f32 -> f16) ----
  k_pad_convert<<<gsz(288 * 512, TPB), TPB, 0, stream>>>(enc_w1, 264, 512, W1p, 288, 512);
  k_pad_convert<<<gsz(512 * 256, TPB), TPB, 0, stream>>>(enc_w2, 512, 256, W2p, 512, 256);
  k_pad_convert<<<gsz(256 * 32, TPB), TPB, 0, stream>>>(enc_w3, 256, 32, W3p, 256, 32);
  k_pad_convert<<<gsz(32 * 256, TPB), TPB, 0, stream>>>(g_w1, 32, 256, GW1p, 32, 256);
  k_pad_convert<<<gsz(256 * 256, TPB), TPB, 0, stream>>>(g_w2, 256, 256, GW2p, 256, 256);
  k_pad_convert<<<gsz(256 * 16, TPB), TPB, 0, stream>>>(g_w3, 256, 4, GW3p, 256, 16);
  for (int e = 0; e < 4; ++e) {
    k_pad_convert<<<gsz(192 * 256, TPB), TPB, 0, stream>>>(
        x_w1 + (size_t)e * 164 * 256, 164, 256, XW1p + (size_t)e * 192 * 256, 192, 256);
    k_pad_convert<<<gsz(256 * 256, TPB), TPB, 0, stream>>>(
        x_w2 + (size_t)e * 256 * 256, 256, 256, XW2p + (size_t)e * 256 * 256, 256, 256);
    k_pad_convert<<<gsz(256 * 144, TPB), TPB, 0, stream>>>(
        x_w3 + (size_t)e * 256 * 132, 256, 132, XW3p + (size_t)e * 256 * 144, 256, 144);
  }
  k_pad_convert<<<gsz((size_t)MCB * 32, TPB), TPB, 0, stream>>>(codebook, MCB, 32, CBh, MCB, 32);

  // ---- encoder ----
  k_build_x0<<<gsz((size_t)NB * 288, TPB), TPB, 0, stream>>>(obs, nxt, X0h);
  k_gemm<<<dim3(32, NB / 64), 128, 0, stream>>>(X0h, 288, W1p, 512, enc_b1,
                                                bufA, 512, nullptr, 0, 288, 512, 1);
  k_gemm<<<dim3(16, NB / 64), 128, 0, stream>>>(bufA, 512, W2p, 256, enc_b2,
                                                bufB, 256, nullptr, 0, 512, 256, 1);
  k_gemm<<<dim3(2, NB / 64), 128, 0, stream>>>(bufB, 256, W3p, 32, enc_b3,
                                               LATh, 32, out_lori, 32, 256, 32, 0);

  // ---- VQ (both argmins fused over one cross product) ----
  k_row_norm<<<gsz(NB, TPB), TPB, 0, stream>>>(out_lori, 32, xnorm, NB);
  k_row_norm<<<gsz(MCB, TPB), TPB, 0, stream>>>(codebook, 32, ynorm, MCB);
  k_init_keys<<<gsz(NB + MCB, TPB), TPB, 0, stream>>>(rowkey, NB, colkey, MCB);
  k_vq<<<dim3(MCB / 64, NB / 64), 128, 0, stream>>>(LATh, CBh, xnorm, ynorm,
                                                    rowkey, colkey);
  k_finalize_rows<<<gsz(NB, TPB), TPB, 0, stream>>>(rowkey, codebook, obs,
                                                    out_ldec, XEh);
  k_finalize_cols<<<gsz(MCB, TPB), TPB, 0, stream>>>(colkey, out_lori, out_lenc);

  // ---- gating ----  (A = XE[:, :32] == latent_dec, lda=192)
  k_gemm<<<dim3(16, NB / 64), 128, 0, stream>>>(XEh, 192, GW1p, 256, g_b1,
                                                bufB, 256, nullptr, 0, 32, 256, 1);
  k_gemm<<<dim3(16, NB / 64), 128, 0, stream>>>(bufB, 256, GW2p, 256, g_b2,
                                                bufA, 256, nullptr, 0, 256, 256, 1);
  k_gemm<<<dim3(1, NB / 64), 128, 0, stream>>>(bufA, 256, GW3p, 16, g_b3,
                                               nullptr, 0, logits, 4, 256, 4, 0);
  k_softmax4<<<gsz(NB, TPB), TPB, 0, stream>>>(logits, gbuf);

  // ---- blended expert MLP ----
  k_gemm_expert<<<dim3(16, NB / 64), 128, 0, stream>>>(
      XEh, 192, XW1p, 256, 192 * 256, gbuf, x_b1, 256,
      bufB, 256, nullptr, 0, 192, 256, 1);
  k_gemm_expert<<<dim3(16, NB / 64), 128, 0, stream>>>(
      bufB, 256, XW2p, 256, 256 * 256, gbuf, x_b2, 256,
      bufA, 256, nullptr, 0, 256, 256, 1);
  k_gemm_expert<<<dim3(9, NB / 64), 128, 0, stream>>>(
      bufA, 256, XW3p, 144, 256 * 144, gbuf, x_b3, 132,
      nullptr, 0, out_output, 132, 256, 132, 0);
}
